// CannyEdge_88673894793673
// MI455X (gfx1250) — matrix operand order repaired and checked by reading, not compiled, not run on hardware
//
#include <hip/hip_runtime.h>
#include <cmath>

typedef __attribute__((ext_vector_type(16))) _Float16 v16h;
typedef __attribute__((ext_vector_type(4)))  _Float16 v4h;
typedef __attribute__((ext_vector_type(8)))  float    v8f;
typedef __attribute__((ext_vector_type(4)))  unsigned int v4u;
typedef __attribute__((ext_vector_type(8)))  int      v8i;
typedef __attribute__((ext_vector_type(4)))  int      v4i;

#define IMGS 32
#define HH 512
#define WW 512
#define HWPIX (HH*WW)

__device__ __forceinline__ int reflect512(int i) { i = (i < 0) ? -i : i; return (i >= 512) ? (1022 - i) : i; }
__device__ __forceinline__ int clamp512(int i)   { return i < 0 ? 0 : (i > 511 ? 511 : i); }

__device__ __forceinline__ float tap5(int d, float g0, float g1, float g2, float g3, float g4) {
  float w = 0.f;
  w = (d == 0) ? g0 : w;
  w = (d == 1) ? g1 : w;
  w = (d == 2) ? g2 : w;
  w = (d == 3) ? g3 : w;
  w = (d == 4) ? g4 : w;
  return w;
}

// Issue a 2D TDM tile load: rows x cols (f16) from global (row stride 512 elems) into LDS,
// rows packed contiguously at stride = cols elements. Caller: wave 0 only, then barrier.
__device__ __forceinline__ void tdm_load_tile_f16(const _Float16* gsrc, void* lds_dst,
                                                  int rows, int cols) {
#if __has_builtin(__builtin_amdgcn_tensor_load_to_lds)
  unsigned long long ga = (unsigned long long)gsrc;
  unsigned lds_off = (unsigned)(unsigned long long)lds_dst;   // low 32 bits of flat LDS addr = LDS byte offset
  v4u g0v;
  g0v[0] = 1u;                                                // count=1, user descriptor
  g0v[1] = lds_off;                                           // lds_addr
  g0v[2] = (unsigned)(ga & 0xFFFFFFFFull);                    // global_addr[31:0]
  g0v[3] = (unsigned)((ga >> 32) & 0x01FFFFFFull) | (2u << 30); // global_addr[56:32] | type=2
  v8i g1v;
  g1v[0] = 1 << 16;                                           // workgroup_mask=0, data_size=1 (2 bytes)
  g1v[1] = (cols & 0xFFFF) << 16;                             // tensor_dim0[15:0]
  g1v[2] = (rows & 0xFFFF) << 16;                             // tensor_dim0 hi=0 | tensor_dim1[15:0]
  g1v[3] = (cols & 0xFFFF) << 16;                             // tensor_dim1 hi=0 | tile_dim0
  g1v[4] = rows & 0xFFFF;                                     // tile_dim1 | tile_dim2=0
  g1v[5] = 512;                                               // tensor_dim0_stride = image row pitch
  g1v[6] = 0;
  g1v[7] = 0;
  v4i z4 = {0, 0, 0, 0};
#if defined(__clang_major__) && (__clang_major__ >= 23)
  v8i z8 = {0, 0, 0, 0, 0, 0, 0, 0};
  __builtin_amdgcn_tensor_load_to_lds(g0v, g1v, z4, z4, z8, 0);
#else
  __builtin_amdgcn_tensor_load_to_lds(g0v, g1v, z4, z4, 0);
#endif
  __builtin_amdgcn_s_wait_tensorcnt(0);
#else
  (void)gsrc; (void)lds_dst; (void)rows; (void)cols;
#endif
}

// ---------------- grayscale: f32 NCHW -> f16 ----------------
__global__ __launch_bounds__(256)
void gray_kernel(const float* __restrict__ x, _Float16* __restrict__ g) {
  size_t tid = (size_t)blockIdx.x * 256 + threadIdx.x;   // one thread per 4 pixels
  size_t p4 = tid * 4;
  if (p4 >= (size_t)IMGS * HWPIX) return;
  size_t img = p4 / HWPIX;
  size_t p = p4 - img * HWPIX;
  const float4 r  = *(const float4*)(x + img * 3 * HWPIX + p);
  const float4 gg = *(const float4*)(x + img * 3 * HWPIX + HWPIX + p);
  const float4 b  = *(const float4*)(x + img * 3 * HWPIX + 2 * HWPIX + p);
  v4h o;
  o[0] = (_Float16)(0.299f * r.x + 0.587f * gg.x + 0.114f * b.x);
  o[1] = (_Float16)(0.299f * r.y + 0.587f * gg.y + 0.114f * b.y);
  o[2] = (_Float16)(0.299f * r.z + 0.587f * gg.z + 0.114f * b.z);
  o[3] = (_Float16)(0.299f * r.w + 0.587f * gg.w + 0.114f * b.w);
  *(v4h*)(g + img * HWPIX + p) = o;
}

// ---------------- horizontal 5-tap blur: TDM tile -> LDS -> WMMA banded GEMM ----------------
// Block = 8 waves = 16 rows x 128 cols of output. Per wave: out(16x16) = A(16x32 image) x B(32x16 banded).
__global__ __launch_bounds__(256)
void blur_h_wmma(const _Float16* __restrict__ in, _Float16* __restrict__ out,
                 float g0, float g1, float g2, float g3, float g4) {
  __shared__ _Float16 band[16 * 132];
  int blk  = blockIdx.x;
  int cs   = (blk & 3) * 128;
  int row0 = ((blk >> 2) & 31) * 16;
  int img  = blk >> 7;
  const _Float16* base = in + (size_t)img * HWPIX;

  int c0  = (cs - 2 < 0) ? 0 : cs - 2;
  int c1  = (cs + 130 > 512) ? 512 : cs + 130;
  int wid = c1 - c0;                      // 130 at image edges, 132 interior

#if __has_builtin(__builtin_amdgcn_tensor_load_to_lds)
  if ((threadIdx.x >> 5) == 0)
    tdm_load_tile_f16(base + (size_t)row0 * WW + c0, band, 16, wid);
#else
  for (int idx = threadIdx.x; idx < 16 * wid; idx += 256) {
    int y = idx / wid, xx = idx - y * wid;
    band[idx] = base[(size_t)(row0 + y) * WW + c0 + xx];
  }
#endif
  __syncthreads();

  int w    = threadIdx.x >> 5;            // wave -> tile column
  int lane = threadIdx.x & 31;
  int lhi  = lane >> 4, llo = lane & 15;
  int col0 = cs + 16 * w;

  v16h a, b;
#pragma unroll
  for (int h = 0; h < 16; ++h) {          // A: M=llo, K per ISA 16-bit A layout
    int k = ((h & 8) << 1) + (h & 7) + 8 * lhi;
    int col = reflect512(col0 - 2 + k);   // reflection targets always live inside the band
    a[h] = band[llo * wid + (col - c0)];  // K>=20 tail multiplies zero weights (finite garbage OK)
  }
#pragma unroll
  for (int h = 0; h < 16; ++h) {          // B: N=llo, K per ISA 16-bit layout (same as A); B[k][n]=g[k-n]
    int k = ((h & 8) << 1) + (h & 7) + 8 * lhi;
    b[h] = (_Float16)tap5(k - llo, g0, g1, g2, g3, g4);
  }
  v8f c = {};
  c = __builtin_amdgcn_wmma_f32_16x16x32_f16(false, a, false, b, (short)0, c, false, false);

  _Float16* ob = out + (size_t)img * HWPIX;
#pragma unroll
  for (int r = 0; r < 8; ++r)             // D: M = r + 8*lhi, N = llo
    ob[(size_t)(row0 + r + 8 * lhi) * WW + col0 + llo] = (_Float16)c[r];
}

// ---------------- vertical 5-tap blur: TDM tile -> LDS -> WMMA banded GEMM ----------------
// Block = 8 waves = 16 rows x 128 cols of output; band = up to 20 source rows x 128 cols.
__global__ __launch_bounds__(256)
void blur_v_wmma(const _Float16* __restrict__ in, _Float16* __restrict__ out,
                 float g0, float g1, float g2, float g3, float g4) {
  __shared__ _Float16 band[20 * 128];
  int blk  = blockIdx.x;
  int cs   = (blk & 3) * 128;
  int row0 = ((blk >> 2) & 31) * 16;
  int img  = blk >> 7;
  const _Float16* base = in + (size_t)img * HWPIX;

  int h0  = (row0 - 2 < 0) ? 0 : row0 - 2;
  int h1  = (row0 + 18 > 512) ? 512 : row0 + 18;
  int hgt = h1 - h0;                      // 18 at image edges, 20 interior

#if __has_builtin(__builtin_amdgcn_tensor_load_to_lds)
  if ((threadIdx.x >> 5) == 0)
    tdm_load_tile_f16(base + (size_t)h0 * WW + cs, band, hgt, 128);
#else
  for (int idx = threadIdx.x; idx < hgt * 128; idx += 256) {
    int y = idx >> 7, xx = idx & 127;
    band[idx] = base[(size_t)(h0 + y) * WW + cs + xx];
  }
#endif
  __syncthreads();

  int w    = threadIdx.x >> 5;
  int lane = threadIdx.x & 31;
  int lhi  = lane >> 4, llo = lane & 15;
  int col0 = cs + 16 * w;

  v16h a, b;
#pragma unroll
  for (int h = 0; h < 16; ++h) {          // A: coefficients, M=llo, A[m][k]=g[k-m]
    int k = ((h & 8) << 1) + (h & 7) + 8 * lhi;
    a[h] = (_Float16)tap5(k - llo, g0, g1, g2, g3, g4);
  }
#pragma unroll
  for (int h = 0; h < 16; ++h) {          // B: K per ISA 16-bit layout (same as A), N = llo
    int k = ((h & 8) << 1) + (h & 7) + 8 * lhi;
    int row = reflect512(row0 - 2 + k);   // K>=20 tail has zero A-weights
    b[h] = band[(row - h0) * 128 + 16 * w + llo];
  }
  v8f c = {};
  c = __builtin_amdgcn_wmma_f32_16x16x32_f16(false, a, false, b, (short)0, c, false, false);

  _Float16* ob = out + (size_t)img * HWPIX;
#pragma unroll
  for (int r = 0; r < 8; ++r)
    ob[(size_t)(row0 + r + 8 * lhi) * WW + col0 + llo] = (_Float16)c[r];
}

// ---------------- fused Sobel + NMS + double threshold ----------------
__device__ __forceinline__ void sobelAt(const _Float16* __restrict__ b, int y, int x,
                                        float& gx, float& gy) {
  int ym = clamp512(y - 1), yp = clamp512(y + 1);
  int xm = clamp512(x - 1), xp = clamp512(x + 1);
  float p00 = (float)b[(size_t)ym * WW + xm], p01 = (float)b[(size_t)ym * WW + x], p02 = (float)b[(size_t)ym * WW + xp];
  float p10 = (float)b[(size_t)y  * WW + xm],                                       p12 = (float)b[(size_t)y  * WW + xp];
  float p20 = (float)b[(size_t)yp * WW + xm], p21 = (float)b[(size_t)yp * WW + x], p22 = (float)b[(size_t)yp * WW + xp];
  gx = (p02 + 2.f * p12 + p22) - (p00 + 2.f * p10 + p20);
  gy = (p20 + 2.f * p21 + p22) - (p00 + 2.f * p01 + p02);
}
__device__ __forceinline__ float magAt(const _Float16* __restrict__ b, int y, int x) {
  float gx, gy; sobelAt(b, y, x, gx, gy);
  return sqrtf(gx * gx + gy * gy + 1e-6f);
}
__device__ __forceinline__ void nbOff(int k, int& dr, int& dc) {
  // octant k -> (dr,dc): 0:(0,1) 1:(1,1) 2:(1,0) 3:(1,-1) 4:(0,-1) 5:(-1,-1) 6:(-1,0) 7:(-1,1)
  dr = (k >= 1 && k <= 3) ? 1 : ((k >= 5) ? -1 : 0);
  dc = (k == 7 || k <= 1) ? 1 : ((k >= 3 && k <= 5) ? -1 : 0);
}

__global__ __launch_bounds__(256)
void sobel_nms_kernel(const _Float16* __restrict__ blr, unsigned char* __restrict__ st) {
  int tid = blockIdx.x * 256 + threadIdx.x;
  if (tid >= IMGS * HWPIX) return;
  int img = tid / HWPIX;
  int p = tid - img * HWPIX;
  int y = p / WW, x = p - y * WW;
  const _Float16* base = blr + (size_t)img * HWPIX;

  float gx, gy; sobelAt(base, y, x, gx, gy);
  float mag = sqrtf(gx * gx + gy * gy + 1e-6f);
  float deg = atan2f(gy, gx) * 57.29577951308232f;
  int o = (int)rintf(deg * (1.0f / 45.0f));        // RNE like jnp.round
  int pos = (o + 8) & 7, neg = (pos + 4) & 7;

  int dr, dc;
  nbOff(pos, dr, dc);
  int yy = y + dr, xx = x + dc;
  float mp = (yy >= 0 && yy < HH && xx >= 0 && xx < WW) ? magAt(base, yy, xx) : 0.f; // SAME conv zero-pads mag
  nbOff(neg, dr, dc);
  yy = y + dr; xx = x + dc;
  float mn = (yy >= 0 && yy < HH && xx >= 0 && xx < WW) ? magAt(base, yy, xx) : 0.f;

  bool keep = (mag - mp > 0.f) && (mag - mn > 0.f);
  float m = keep ? mag : 0.f;
  st[tid] = (m > 0.15f) ? (unsigned char)2 : ((m > 0.05f) ? (unsigned char)1 : (unsigned char)0);
}

// ---------------- hysteresis: in-LDS relaxation, ping-pong global passes ----------------
__global__ __launch_bounds__(256)
void hyst_kernel(const unsigned char* __restrict__ sin_, unsigned char* __restrict__ sout) {
  __shared__ unsigned char s[18][18];
  size_t imgoff = (size_t)blockIdx.z * HWPIX;
  int lin = threadIdx.y * 16 + threadIdx.x;
  for (int idx = lin; idx < 18 * 18; idx += 256) {
    int rr = idx / 18, cc = idx % 18;
    int gy = blockIdx.y * 16 + rr - 1;
    int gx = blockIdx.x * 16 + cc - 1;
    unsigned char v = 0;                                  // zero-pad: OOB is never strong
    if (gy >= 0 && gy < HH && gx >= 0 && gx < WW) v = sin_[imgoff + (size_t)gy * WW + gx];
    s[rr][cc] = v;
  }
  __syncthreads();

  int r = threadIdx.y + 1, c = threadIdx.x + 1;
#pragma unroll 1
  for (int it = 0; it < 16; ++it) {
    unsigned char v = s[r][c];
    unsigned char nv = v;
    if (v == 1) {
      bool strong = s[r-1][c-1] == 2 || s[r-1][c] == 2 || s[r-1][c+1] == 2 ||
                    s[r  ][c-1] == 2 ||                    s[r  ][c+1] == 2 ||
                    s[r+1][c-1] == 2 || s[r+1][c] == 2 || s[r+1][c+1] == 2;
      if (strong) nv = 2;
    }
    __syncthreads();
    s[r][c] = nv;
    __syncthreads();
  }
  int gy = blockIdx.y * 16 + threadIdx.y;
  int gx = blockIdx.x * 16 + threadIdx.x;
  sout[imgoff + (size_t)gy * WW + gx] = s[r][c];
}

__global__ __launch_bounds__(256)
void finalize_kernel(const unsigned char* __restrict__ st, float* __restrict__ out) {
  int tid = blockIdx.x * 256 + threadIdx.x;
  if (tid >= IMGS * HWPIX) return;
  out[tid] = (st[tid] == 2) ? 1.0f : 0.0f;
}

// ---------------- host ----------------
static void gauss5(float sigma, float* g) {
  float s = 0.f;
  for (int i = 0; i < 5; ++i) { float t = (float)(i - 2); g[i] = expf(-t * t / (2.f * sigma * sigma)); s += g[i]; }
  for (int i = 0; i < 5; ++i) g[i] /= s;
}

extern "C" void kernel_launch(void* const* d_in, const int* in_sizes, int n_in,
                              void* d_out, int out_size, void* d_ws, size_t ws_size,
                              hipStream_t stream) {
  (void)in_sizes; (void)n_in; (void)out_size; (void)ws_size;
  const float* x = (const float*)d_in[0];
  float* out = (float*)d_out;
  const size_t N = (size_t)IMGS * HWPIX;

  char* ws = (char*)d_ws;
  _Float16* gray = (_Float16*)(ws);                   // 2N bytes
  _Float16* bufA = (_Float16*)(ws + 2 * N);           // 2N bytes
  _Float16* bufB = (_Float16*)(ws + 4 * N);           // 2N bytes
  unsigned char* stA = (unsigned char*)(ws + 6 * N);  // N bytes
  unsigned char* stB = (unsigned char*)(ws + 7 * N);  // N bytes

  float gA[5], gB[5];
  gauss5(1.5f, gA);
  gauss5(1.0f, gB);

  int grayBlocks = (int)((N / 4 + 255) / 256);
  gray_kernel<<<grayBlocks, 256, 0, stream>>>(x, gray);

  int blurBlocks = IMGS * (HH / 16) * (WW / 128);     // 4096 blocks, 8 waves each
  blur_h_wmma<<<blurBlocks, 256, 0, stream>>>(gray, bufA, gA[0], gA[1], gA[2], gA[3], gA[4]);
  blur_v_wmma<<<blurBlocks, 256, 0, stream>>>(bufA, bufB, gA[0], gA[1], gA[2], gA[3], gA[4]);
  blur_h_wmma<<<blurBlocks, 256, 0, stream>>>(bufB, gray, gB[0], gB[1], gB[2], gB[3], gB[4]);
  blur_v_wmma<<<blurBlocks, 256, 0, stream>>>(gray, bufA, gB[0], gB[1], gB[2], gB[3], gB[4]);

  int pixBlocks = (int)((N + 255) / 256);
  sobel_nms_kernel<<<pixBlocks, 256, 0, stream>>>(bufA, stA);

  dim3 hgrid(WW / 16, HH / 16, IMGS), hblk(16, 16);
  for (int i = 0; i < 24; ++i) {                      // fixed, deterministic schedule
    const unsigned char* si = (i & 1) ? stB : stA;
    unsigned char* so       = (i & 1) ? stA : stB;
    hyst_kernel<<<hgrid, hblk, 0, stream>>>(si, so);
  }
  finalize_kernel<<<pixBlocks, 256, 0, stream>>>(stA, out);  // 24 passes end back in stA
}
